// LatentStateImitation_8315056685701
// MI455X (gfx1250) — matrix operand order repaired and checked by reading, not compiled, not run on hardware
//
#include <hip/hip_runtime.h>
#include <hip/hip_bf16.h>

// Problem constants (from reference)
#define BB      256
#define TT      1024
#define SDIM    64
#define LATENT  32
#define HH      128
#define ADIM    8
#define NMIX    3
#define IN_DIM  (SDIM + LATENT)      // 96
#define KTOT    (IN_DIM + HH)        // 224  (x | h concat K-dim)
#define GG      (4 * HH)             // 512 gate columns
#define NT_GATE (GG / 16)            // 32 gate N-tiles
#define KC      (KTOT / 32)          // 7 K-chunks of 32
#define KCX     (IN_DIM / 32)        // 3 x-chunks
#define KCH     (HH / 32)            // 4 h-chunks
#define MU_N    (NMIX * ADIM)        // 24
#define HB      8                    // head batch (steps)

typedef __attribute__((ext_vector_type(16))) __bf16 v16bf;
typedef __attribute__((ext_vector_type(8)))  float  v8f;

#if __has_builtin(__builtin_amdgcn_tanhf)
__device__ __forceinline__ float tanh_hw(float x) { return __builtin_amdgcn_tanhf(x); }
#else
__device__ __forceinline__ float tanh_hw(float x) { return tanhf(x); }
#endif
// sigmoid(x) = 0.5*tanh(0.5x) + 0.5  (one v_tanh_f32 + one fma)
__device__ __forceinline__ float sig_hw(float x) { return fmaf(0.5f, tanh_hw(0.5f * x), 0.5f); }

__launch_bounds__(256, 1)
__global__ void lstm_mdn_fused(
    const float* __restrict__ states,   // [B,T,SDIM]
    const float* __restrict__ lat,      // [B,LATENT]
    const float* __restrict__ W_ih,     // [4H, IN_DIM]
    const float* __restrict__ W_hh,     // [4H, H]
    const float* __restrict__ b_ih,     // [4H]
    const float* __restrict__ b_hh,     // [4H]
    const float* __restrict__ W_mu,     // [H, MU_N]
    const float* __restrict__ b_mu,     // [MU_N]
    const float* __restrict__ W_sg,     // [H, NMIX]
    const float* __restrict__ b_sg,     // [NMIX]
    const float* __restrict__ W_al,     // [H, NMIX]
    const float* __restrict__ b_al,     // [NMIX]
    float* __restrict__ out)            // mu | sigma_inv | alpha (flat, f32)
{
    // ---- LDS (~312.5 KB of the 320 KB/WGP) ----
    __shared__ __bf16 Wl   [NT_GATE * KC * 32 * 16]; // 224 KB gate weights, B-frag layout
    __shared__ __bf16 Wmul [2 * KCH * 32 * 16];      // 8 KB   mu head weights
    __shared__ __bf16 Wsal [KCH * 32 * 16];          // 4 KB   sig|alpha head weights
    __shared__ __bf16 xbuf [2][16][IN_DIM];          // 6 KB   double-buffered x tile (bf16)
    __shared__ __bf16 hbuf [16][HH];                 // 4 KB   current h tile (A-matrix)
    __shared__ __bf16 hhist[HB][16][HH];             // 32 KB  h history ring for batched heads
    __shared__ float  gbuf [16][GG + 4];             // 33 KB  gate pre-activations (padded)
    __shared__ float  bsum [GG];                     // b_ih + b_hh
    __shared__ float  bmu  [MU_N];
    __shared__ float  bsa  [2 * NMIX];

    float* sabuf = &gbuf[0][0];                      // head scratch aliases gbuf: [HB][16][8]

    const int tid   = threadIdx.x;
    const int lane  = tid & 31;
    const int wave  = tid >> 5;
    const int row0  = blockIdx.x * 16;               // batch tile base row
    const int arow  = lane & 15;                     // A-frag row for this lane
    const int ahalf = (lane < 16) ? 0 : 8;           // A-frag K sub-offset / D-frag M offset

    // ================= one-time fills =================
    // Gate weights: Wt[k][n], k=0..95 -> W_ih^T, k=96..223 -> W_hh^T,
// swizzled so each lane holds K = 8h+0..7 and 16+8h+0..7 of its chunk (32 B per lane).
    for (int idx = tid; idx < NT_GATE * KC * 32; idx += 256) {
        const int nt  = idx / (KC * 32);
        const int rem = idx % (KC * 32);
        const int c   = rem >> 5;
        const int l   = rem & 31;
        const int col = nt * 16 + (l & 15);
        const int kb  = c * 32 + ((l < 16) ? 0 : 8);
        __bf16* dst = &Wl[idx * 16];
        #pragma unroll
        for (int e = 0; e < 16; ++e) {
            const int k = kb + ((e < 8) ? e : e + 8);
            const float v = (k < IN_DIM) ? W_ih[col * IN_DIM + k]
                                         : W_hh[col * HH + (k - IN_DIM)];
            dst[e] = (__bf16)v;
        }
    }
    // mu head weights [K=128][N=24 -> pad 32]
    for (int idx = tid; idx < 2 * KCH * 32; idx += 256) {
        const int nt  = idx / (KCH * 32);
        const int rem = idx % (KCH * 32);
        const int c   = rem >> 5;
        const int l   = rem & 31;
        const int n   = nt * 16 + (l & 15);
        const int kb  = c * 32 + ((l < 16) ? 0 : 8);
        __bf16* dst = &Wmul[idx * 16];
        #pragma unroll
        for (int e = 0; e < 16; ++e) {
            const int k = kb + ((e < 8) ? e : e + 8);
            dst[e] = (__bf16)((n < MU_N) ? W_mu[k * MU_N + n] : 0.0f);
        }
    }
    // sig(0..2) | alpha(3..5) head weights [K=128][N=6 -> pad 16]
    for (int idx = tid; idx < KCH * 32; idx += 256) {
        const int c  = idx >> 5;
        const int l  = idx & 31;
        const int n  = l & 15;
        const int kb = c * 32 + ((l < 16) ? 0 : 8);
        __bf16* dst = &Wsal[idx * 16];
        #pragma unroll
        for (int e = 0; e < 16; ++e) {
            const int k = kb + ((e < 8) ? e : e + 8);
            float v = 0.0f;
            if (n < NMIX)          v = W_sg[k * NMIX + n];
            else if (n < 2 * NMIX) v = W_al[k * NMIX + (n - NMIX)];
            dst[e] = (__bf16)v;
        }
    }
    for (int i = tid; i < GG; i += 256) bsum[i] = b_ih[i] + b_hh[i];
    if (tid < MU_N)                    bmu[tid] = b_mu[tid];
    if (tid < NMIX)                    bsa[tid] = b_sg[tid];
    if (tid >= NMIX && tid < 2 * NMIX) bsa[tid] = b_al[tid - NMIX];
    // latent columns (constant over t) into BOTH x buffers; h0 = 0
    for (int i = tid; i < 16 * LATENT; i += 256) {
        const int r = i >> 5, c = i & 31;
        const __bf16 v = (__bf16)lat[(size_t)(row0 + r) * LATENT + c];
        xbuf[0][r][SDIM + c] = v;
        xbuf[1][r][SDIM + c] = v;
    }
    for (int i = tid; i < 16 * HH; i += 256) {
        const int r = i >> 7, c = i & 127;
        hbuf[r][c] = (__bf16)0.0f;
    }
    // preload x for t = 0
    {
        const int r  = tid >> 4;
        const int c4 = (tid & 15) * 4;
        const float4 v = *(const float4*)(states + ((size_t)(row0 + r) * TT) * SDIM + c4);
        xbuf[0][r][c4 + 0] = (__bf16)v.x;
        xbuf[0][r][c4 + 1] = (__bf16)v.y;
        xbuf[0][r][c4 + 2] = (__bf16)v.z;
        xbuf[0][r][c4 + 3] = (__bf16)v.w;
    }

    // cell state: each thread owns 8 contiguous h-columns of one row
    const int r_c = tid >> 4;
    const int h0  = (tid & 15) * 8;
    float cc[8];
    #pragma unroll
    for (int e = 0; e < 8; ++e) cc[e] = 0.0f;

    const size_t sig_base = (size_t)BB * TT * MU_N;
    const size_t al_base  = sig_base + (size_t)BB * TT * NMIX;

    __syncthreads();

    // ================= scan over T =================
    for (int t = 0; t < TT; ++t) {
        const int par = t & 1;

        // ---- stage next step's x tile into the other buffer (hides under WMMAs) ----
        if (t + 1 < TT) {
            const int r  = tid >> 4;
            const int c4 = (tid & 15) * 4;
            const float* p = states + ((size_t)(row0 + r) * TT + (t + 1)) * SDIM + c4;
            const float4 v = *(const float4*)p;
            __bf16* q = &xbuf[par ^ 1][r][c4];
            q[0] = (__bf16)v.x; q[1] = (__bf16)v.y; q[2] = (__bf16)v.z; q[3] = (__bf16)v.w;
            if (t + 2 < TT) __builtin_prefetch(p + SDIM, 0, 0);
        }

        // ---- gate GEMM: [16,224]x[224,512] via bf16 WMMA ----
        {
            v8f acc[4];
            #pragma unroll
            for (int nt = 0; nt < 4; ++nt) acc[nt] = v8f{};
            // x chunks (k = 0..95)
            #pragma unroll
            for (int c = 0; c < KCX; ++c) {
                v16bf a;
                const int klo = c * 32 + ahalf;
                ((uint4*)&a)[0] = *(const uint4*)&xbuf[par][arow][klo];
                ((uint4*)&a)[1] = *(const uint4*)&xbuf[par][arow][klo + 16];
                #pragma unroll
                for (int nt = 0; nt < 4; ++nt) {
                    v16bf bfr;
                    const __bf16* src = &Wl[(((wave * 4 + nt) * KC + c) * 32 + lane) * 16];
                    ((uint4*)&bfr)[0] = ((const uint4*)src)[0];
                    ((uint4*)&bfr)[1] = ((const uint4*)src)[1];
                    acc[nt] = __builtin_amdgcn_wmma_f32_16x16x32_bf16(
                        false, a, false, bfr, (short)0, acc[nt], false, false);
                }
            }
            // h chunks (k = 96..223)
            #pragma unroll
            for (int c = KCX; c < KC; ++c) {
                v16bf a;
                const int klo = (c - KCX) * 32 + ahalf;
                ((uint4*)&a)[0] = *(const uint4*)&hbuf[arow][klo];
                ((uint4*)&a)[1] = *(const uint4*)&hbuf[arow][klo + 16];
                #pragma unroll
                for (int nt = 0; nt < 4; ++nt) {
                    v16bf bfr;
                    const __bf16* src = &Wl[(((wave * 4 + nt) * KC + c) * 32 + lane) * 16];
                    ((uint4*)&bfr)[0] = ((const uint4*)src)[0];
                    ((uint4*)&bfr)[1] = ((const uint4*)src)[1];
                    acc[nt] = __builtin_amdgcn_wmma_f32_16x16x32_bf16(
                        false, a, false, bfr, (short)0, acc[nt], false, false);
                }
            }
            #pragma unroll
            for (int nt = 0; nt < 4; ++nt) {
                const int col = wave * 64 + nt * 16 + (lane & 15);
                const float bb = bsum[col];
                #pragma unroll
                for (int v = 0; v < 8; ++v)
                    gbuf[v + ahalf][col] = acc[nt][v] + bb;
            }
        }
        __syncthreads();   // gates ready; staged x visible for next step

        // ---- elementwise cell update (v_tanh_f32 path) ----
        {
            __bf16* hrow  = &hbuf[r_c][h0];
            __bf16* hhrow = &hhist[t & (HB - 1)][r_c][h0];
            #pragma unroll
            for (int e = 0; e < 8; ++e) {
                const int hc = h0 + e;
                const float ig = sig_hw(gbuf[r_c][hc]);
                const float fg = sig_hw(gbuf[r_c][HH + hc]);
                const float gg = tanh_hw(gbuf[r_c][2 * HH + hc]);
                const float og = sig_hw(gbuf[r_c][3 * HH + hc]);
                cc[e] = fg * cc[e] + ig * gg;
                const __bf16 h = (__bf16)(og * tanh_hw(cc[e]));
                hrow[e]  = h;
                hhrow[e] = h;
            }
        }
        __syncthreads();   // h ready for next recurrence / head batch

        // ---- batched MDN heads: every HB steps, wave w handles history slot w ----
        if ((t & (HB - 1)) == (HB - 1)) {
            const int s  = wave;                 // history slot (= 8 waves)
            const int ts = t - (HB - 1) + s;     // timestep of this slot
            v8f am0 = v8f{}, am1 = v8f{}, asa = v8f{};
            #pragma unroll
            for (int c = 0; c < KCH; ++c) {
                v16bf a;
                const int klo = c * 32 + ahalf;
                ((uint4*)&a)[0] = *(const uint4*)&hhist[s][arow][klo];
                ((uint4*)&a)[1] = *(const uint4*)&hhist[s][arow][klo + 16];
                v16bf b0, b1, b2;
                {
                    const __bf16* p0 = &Wmul[((0 * KCH + c) * 32 + lane) * 16];
                    ((uint4*)&b0)[0] = ((const uint4*)p0)[0];
                    ((uint4*)&b0)[1] = ((const uint4*)p0)[1];
                    const __bf16* p1 = &Wmul[((1 * KCH + c) * 32 + lane) * 16];
                    ((uint4*)&b1)[0] = ((const uint4*)p1)[0];
                    ((uint4*)&b1)[1] = ((const uint4*)p1)[1];
                    const __bf16* p2 = &Wsal[(c * 32 + lane) * 16];
                    ((uint4*)&b2)[0] = ((const uint4*)p2)[0];
                    ((uint4*)&b2)[1] = ((const uint4*)p2)[1];
                }
                am0 = __builtin_amdgcn_wmma_f32_16x16x32_bf16(false, a, false, b0, (short)0, am0, false, false);
                am1 = __builtin_amdgcn_wmma_f32_16x16x32_bf16(false, a, false, b1, (short)0, am1, false, false);
                asa = __builtin_amdgcn_wmma_f32_16x16x32_bf16(false, a, false, b2, (short)0, asa, false, false);
            }
            // mu -> global
            {
                const int j0 = lane & 15;
                if (j0 < MU_N) {  // tile 0 cols 0..15 always valid
                    const float bb = bmu[j0];
                    #pragma unroll
                    for (int v = 0; v < 8; ++v)
                        out[((size_t)(row0 + v + ahalf) * TT + ts) * MU_N + j0] = am0[v] + bb;
                }
                const int j1 = 16 + j0;
                if (j1 < MU_N) {
                    const float bb = bmu[j1];
                    #pragma unroll
                    for (int v = 0; v < 8; ++v)
                        out[((size_t)(row0 + v + ahalf) * TT + ts) * MU_N + j1] = am1[v] + bb;
                }
            }
            // sig|alpha pre-activations -> scratch (aliases gbuf, free here)
            {
                const int n = lane & 15;
                if (n < 8) {
                    #pragma unroll
                    for (int v = 0; v < 8; ++v)
                        sabuf[((s * 16) + (v + ahalf)) * 8 + n] = asa[v];
                }
            }
            __syncthreads();
            // finalize sigma_inv (exp) and alpha (softmax over 3): 128 threads
            if (tid < 16 * HB) {
                const int s2 = tid >> 4;
                const int r  = tid & 15;
                const int t2 = t - (HB - 1) + s2;
                const float* sa = &sabuf[((s2 * 16) + r) * 8];
                const size_t o3 = ((size_t)(row0 + r) * TT + t2) * NMIX;
                out[sig_base + o3 + 0] = __expf(sa[0] + bsa[0]);
                out[sig_base + o3 + 1] = __expf(sa[1] + bsa[1]);
                out[sig_base + o3 + 2] = __expf(sa[2] + bsa[2]);
                const float a0 = sa[3] + bsa[3];
                const float a1 = sa[4] + bsa[4];
                const float a2 = sa[5] + bsa[5];
                const float mx = fmaxf(a0, fmaxf(a1, a2));
                const float e0 = __expf(a0 - mx), e1 = __expf(a1 - mx), e2 = __expf(a2 - mx);
                const float inv = 1.0f / (e0 + e1 + e2);
                out[al_base + o3 + 0] = e0 * inv;
                out[al_base + o3 + 1] = e1 * inv;
                out[al_base + o3 + 2] = e2 * inv;
            }
            __syncthreads();  // scratch free before next step's gate writes
        }
    }
}

extern "C" void kernel_launch(void* const* d_in, const int* in_sizes, int n_in,
                              void* d_out, int out_size, void* d_ws, size_t ws_size,
                              hipStream_t stream) {
    (void)in_sizes; (void)n_in; (void)out_size; (void)d_ws; (void)ws_size;
    const float* states = (const float*)d_in[0];
    const float* lat    = (const float*)d_in[1];
    const float* W_ih   = (const float*)d_in[2];
    const float* W_hh   = (const float*)d_in[3];
    const float* b_ih   = (const float*)d_in[4];
    const float* b_hh   = (const float*)d_in[5];
    const float* W_mu   = (const float*)d_in[6];
    const float* b_mu   = (const float*)d_in[7];
    const float* W_sg   = (const float*)d_in[8];
    const float* b_sg   = (const float*)d_in[9];
    const float* W_al   = (const float*)d_in[10];
    const float* b_al   = (const float*)d_in[11];
    float* out = (float*)d_out;

    lstm_mdn_fused<<<dim3(BB / 16), dim3(256), 0, stream>>>(
        states, lat, W_ih, W_hh, b_ih, b_hh,
        W_mu, b_mu, W_sg, b_sg, W_al, b_al, out);
}
